// ImprovedTransformerBlock_60842506715145
// MI455X (gfx1250) — matrix operand order repaired and checked by reading, not compiled, not run on hardware
//
#include <hip/hip_runtime.h>
#include <math.h>

// ---------------------------------------------------------------------------
// CDNA5 (gfx1250) wave32 WMMA bf16 types & helpers
// ---------------------------------------------------------------------------
typedef __attribute__((ext_vector_type(16))) __bf16 bf16x16;
typedef __attribute__((ext_vector_type(8)))  float  f32x8;

static __device__ inline f32x8 wmma_bf16(bf16x16 a, bf16x16 b, f32x8 c) {
  // D = A(16x32) * B(32x16) + C(16x16 f32)
  return __builtin_amdgcn_wmma_f32_16x16x32_bf16(
      /*neg_a=*/false, a, /*neg_b=*/false, b,
      /*c_mod=*/(short)0, c, /*reuse_a=*/false, /*reuse_b=*/false);
}

// raw LDS byte address (addrspace(3)) of a generic pointer into __shared__
typedef __attribute__((address_space(3))) char lds_char;
static __device__ inline unsigned lds_off(const void* p) {
  return (unsigned)(unsigned long long)(lds_char*)p;
}

// CDNA5 async DMA: global -> LDS, 16 bytes per lane, tracked by ASYNCcnt.
#define ASYNC_B128(loff, gaddr)                                      \
  asm volatile("global_load_async_to_lds_b128 %0, %1, off"           \
               :: "v"(loff), "v"(gaddr) : "memory")
#define WAIT_ASYNC0() asm volatile("s_wait_asynccnt 0" ::: "memory")

// A-fragment (16x32 bf16, row-major LDS tile with leading dim `ld`):
// lane 0-15  -> M=lane, K = {0..7, 16..23}
// lane 16-31 -> M=lane-16, K = {8..15, 24..31}
static __device__ inline bf16x16 frag_a(const __bf16* base, int ld) {
  int lane = threadIdx.x & 31;
  int m  = lane & 15;
  int kh = lane >> 4;
  const __bf16* p = base + m * ld + kh * 8;
  union { uint4 u[2]; bf16x16 v; } c;
  c.u[0] = *(const uint4*)(p);       // K = kh*8 + 0..7
  c.u[1] = *(const uint4*)(p + 16);  // K = 16 + kh*8 + 0..7
  return c.v;
}

// B-fragment (32x16 bf16) where the tile is stored [N][K] row-major (ld = K
// stride), i.e. weight layout. lane 0-15 -> K={0..7,16..23}, lane 16-31 -> K={8..15,24..31},
// column N = lane & 15.
static __device__ inline bf16x16 frag_bt(const __bf16* base, int ld) {
  int lane = threadIdx.x & 31;
  int n  = lane & 15;
  int kb = (lane >> 4) * 8;
  const __bf16* p = base + n * ld + kb;
  union { uint4 u[2]; bf16x16 v; } c;
  c.u[0] = *(const uint4*)(p);
  c.u[1] = *(const uint4*)(p + 16);
  return c.v;
}

// ---------------------------------------------------------------------------
// Generic tiled bf16 GEMM:  C(M,N) = A(M,K) @ Bw(N,K)^T  (+ bias) + epilogue
// 256 threads = 8 waves; block tile 128x128, K-step 32; wave tile 64x32.
// ASYNC=true : double-buffered LDS, tiles staged with global_load_async_to_lds
//              (next tile DMA overlaps current tile WMMA).
// ASYNC=false: register-staged path supporting the shifted/zero-padded A
//              gather used by the dilated conv taps.
// MODE 0: out bf16 = acc + bias
// MODE 1: out bf16 = gelu(acc + bias)      (exact erf GELU)
// MODE 2: out f32 += acc + bias            (residual accumulate)
// MODE 5: out f32 += rowscale[m]*(acc+bias)  (MoE gate-weighted residual)
// ---------------------------------------------------------------------------
template<int MODE, bool ASYNC>
__global__ __launch_bounds__(256) void gemm_bf16_kernel(
    const __bf16* __restrict__ A, const __bf16* __restrict__ Bw,
    const float* __restrict__ bias, void* __restrict__ Cout,
    const float* __restrict__ rowscale, int scale_stride, int scale_off,
    int M, int N, int K, int row_shift, int seq_len) {
  constexpr int BM = 128, BN = 128, BK = 32, LDT = 40;
  __shared__ __bf16 As[2][BM * LDT];
  __shared__ __bf16 Bs[2][BN * LDT];
  const int tid = threadIdx.x;
  const int wid = tid >> 5;
  const int wm = wid & 1;         // 2 waves along M (64 rows each)
  const int wn = wid >> 1;        // 4 waves along N (32 cols each)
  const int bm = blockIdx.x * BM;
  const int bn = blockIdx.y * BN;
  // staging coords: each thread owns 16B chunks (r0,col0) and (r0+64,col0)
  const int r0 = tid >> 2;
  const int col0 = (tid & 3) * 8;

  const f32x8 zero = {0.f, 0.f, 0.f, 0.f, 0.f, 0.f, 0.f, 0.f};
  f32x8 acc[4][2];
#pragma unroll
  for (int mt = 0; mt < 4; ++mt)
#pragma unroll
    for (int nt = 0; nt < 2; ++nt) acc[mt][nt] = zero;

  auto compute = [&](const __bf16* Abuf, const __bf16* Bbuf) {
    bf16x16 af[4], bfr[2];
#pragma unroll
    for (int mt = 0; mt < 4; ++mt)
      af[mt] = frag_a(Abuf + (wm * 64 + mt * 16) * LDT, LDT);
#pragma unroll
    for (int nt = 0; nt < 2; ++nt)
      bfr[nt] = frag_bt(Bbuf + (wn * 32 + nt * 16) * LDT, LDT);
#pragma unroll
    for (int mt = 0; mt < 4; ++mt)
#pragma unroll
      for (int nt = 0; nt < 2; ++nt)
        acc[mt][nt] = wmma_bf16(af[mt], bfr[nt], acc[mt][nt]);
  };

  const int nK = K / BK;

  if constexpr (ASYNC) {
    // ---- async double-buffered pipeline (row_shift == 0) ----
    unsigned long long ga0 =
        (unsigned long long)(A + (size_t)(bm + r0) * K + col0);
    unsigned long long ga1 =
        (unsigned long long)(A + (size_t)(bm + r0 + 64) * K + col0);
    unsigned long long gb0 =
        (unsigned long long)(Bw + (size_t)(bn + r0) * K + col0);
    unsigned long long gb1 =
        (unsigned long long)(Bw + (size_t)(bn + r0 + 64) * K + col0);
    unsigned la0[2], la1[2], lb0[2], lb1[2];
#pragma unroll
    for (int b = 0; b < 2; ++b) {
      la0[b] = lds_off(&As[b][r0 * LDT + col0]);
      la1[b] = lds_off(&As[b][(r0 + 64) * LDT + col0]);
      lb0[b] = lds_off(&Bs[b][r0 * LDT + col0]);
      lb1[b] = lds_off(&Bs[b][(r0 + 64) * LDT + col0]);
    }
    // prologue: DMA tile 0 into buffer 0
    ASYNC_B128(la0[0], ga0); ASYNC_B128(la1[0], ga1);
    ASYNC_B128(lb0[0], gb0); ASYNC_B128(lb1[0], gb1);
    ga0 += BK * 2; ga1 += BK * 2; gb0 += BK * 2; gb1 += BK * 2;
    WAIT_ASYNC0();
    __syncthreads();
    for (int kb = 0; kb < nK; ++kb) {
      const int cur = kb & 1;
      if (kb + 1 < nK) {
        const int nxt = cur ^ 1;
        ASYNC_B128(la0[nxt], ga0); ASYNC_B128(la1[nxt], ga1);
        ASYNC_B128(lb0[nxt], gb0); ASYNC_B128(lb1[nxt], gb1);
        ga0 += BK * 2; ga1 += BK * 2; gb0 += BK * 2; gb1 += BK * 2;
      }
      compute(&As[cur][0], &Bs[cur][0]);
      WAIT_ASYNC0();      // next tile landed in LDS
      __syncthreads();    // visible to all waves; cur buffer free to overwrite
    }
  } else {
    // ---- register-staged path with shifted/zero-padded A rows (conv) ----
    for (int kb = 0; kb < nK; ++kb) {
      const int k0 = kb * BK;
#pragma unroll
      for (int i = 0; i < 2; ++i) {
        int r = r0 + i * 64;
        uint4 av = make_uint4(0u, 0u, 0u, 0u);
        int gm = bm + r;
        int pos = gm % seq_len + row_shift;
        if (pos >= 0 && pos < seq_len)
          av = *(const uint4*)(A + (size_t)(gm + row_shift) * K + k0 + col0);
        *(uint4*)(&As[0][r * LDT + col0]) = av;
        uint4 bv = *(const uint4*)(Bw + (size_t)(bn + r) * K + k0 + col0);
        *(uint4*)(&Bs[0][r * LDT + col0]) = bv;
      }
      __syncthreads();
      compute(&As[0][0], &Bs[0][0]);
      __syncthreads();
    }
  }

  // epilogue: C/D layout -> lane n = lane&15, rows m = r + (lane>>4)*8
  const int lane = tid & 31;
  const int n_in = lane & 15;
  const int mrow = (lane >> 4) * 8;
#pragma unroll
  for (int mt = 0; mt < 4; ++mt)
#pragma unroll
    for (int nt = 0; nt < 2; ++nt) {
      int gn = bn + wn * 32 + nt * 16 + n_in;
      float bvv = (bias != nullptr) ? bias[gn] : 0.f;
#pragma unroll
      for (int r = 0; r < 8; ++r) {
        int gm = bm + wm * 64 + mt * 16 + mrow + r;
        float v = acc[mt][nt][r] + bvv;
        size_t idx = (size_t)gm * N + gn;
        if constexpr (MODE == 0) {
          ((__bf16*)Cout)[idx] = (__bf16)v;
        } else if constexpr (MODE == 1) {
          v = 0.5f * v * (1.0f + erff(v * 0.70710678118654752f));
          ((__bf16*)Cout)[idx] = (__bf16)v;
        } else if constexpr (MODE == 2) {
          ((float*)Cout)[idx] += v;
        } else if constexpr (MODE == 5) {
          float g = rowscale[(size_t)gm * scale_stride + scale_off];
          ((float*)Cout)[idx] += g * v;
        }
      }
    }
}

// ---------------------------------------------------------------------------
// Windowed attention, one workgroup per (window n<8, head, batch).
// S[64,64] = Q[64,256] K[64,256]^T * 1/8 ; P = softmax(S) ;
// O[64,256] = P V ; scattered per torch-unfold reshape:
//   j = n*256 + d*4 + w/64 ,  col = h*64 + w%64   (j < 2048 always for n<8;
//   windows n>=8 are entirely truncated by o[:, :L])
// Dynamic LDS = 128 KB (Q,K channel-major; V token-major; S f32; P bf16).
// ---------------------------------------------------------------------------
#define ATTN_SMEM 131072

__global__ __launch_bounds__(256) void attn_window_kernel(
    const __bf16* __restrict__ qkv, __bf16* __restrict__ o_flat) {
  constexpr int L = 2048, C3 = 3072, HD = 64;
  constexpr int LDQ = 264, LDV = 72, LDS_S = 68, LDP = 72;
  extern __shared__ char smem[];
  __bf16* qs = (__bf16*)smem;              // [64][LDQ]   channel-major
  __bf16* ks = qs + 64 * LDQ;              // [64][LDQ]   channel-major
  __bf16* vs = ks + 64 * LDQ;              // [256][LDV]  token-major
  float*  ss = (float*)(vs + 256 * LDV);   // [64][LDS_S] scores f32
  __bf16* ps = (__bf16*)(ss + 64 * LDS_S); // [64][LDP]   probs bf16

  const int nwin = blockIdx.x;  // 0..7
  const int h    = blockIdx.y;
  const int b    = blockIdx.z;
  const int tid  = threadIdx.x;
  const int lane = tid & 31;
  const int wid  = tid >> 5;

  // ---- stage window: thread t owns token w=t (transpose Q/K into LDS) ----
  {
    const int w = tid;
    const int l = nwin * 128 + w;
    const __bf16* rowp = qkv + ((size_t)(b * L + l)) * C3 + h * HD;
#pragma unroll
    for (int c = 0; c < 8; ++c) {
      union { uint4 u; __bf16 e[8]; } uq, uk;
      uq.u = *(const uint4*)(rowp + c * 8);
      uk.u = *(const uint4*)(rowp + 1024 + c * 8);
      uint4 uv = *(const uint4*)(rowp + 2048 + c * 8);
#pragma unroll
      for (int j = 0; j < 8; ++j) {
        qs[(c * 8 + j) * LDQ + w] = uq.e[j];
        ks[(c * 8 + j) * LDQ + w] = uk.e[j];
      }
      *(uint4*)(vs + w * LDV + c * 8) = uv;
    }
  }
  __syncthreads();

  // ---- S = Q K^T : 16 tiles of 16x16, 2 per wave, K-dim 256 ----
  {
    const f32x8 zero = {0.f, 0.f, 0.f, 0.f, 0.f, 0.f, 0.f, 0.f};
    f32x8 acc0 = zero, acc1 = zero;
    const int t0 = wid * 2;
    const int mt0 = t0 >> 2, nt0 = t0 & 3;
    const int mt1 = (t0 + 1) >> 2, nt1 = (t0 + 1) & 3;
    for (int k0 = 0; k0 < 256; k0 += 32) {
      bf16x16 a0 = frag_a(qs + (mt0 * 16) * LDQ + k0, LDQ);
      bf16x16 a1 = frag_a(qs + (mt1 * 16) * LDQ + k0, LDQ);
      bf16x16 b0 = frag_bt(ks + (nt0 * 16) * LDQ + k0, LDQ);
      bf16x16 b1 = frag_bt(ks + (nt1 * 16) * LDQ + k0, LDQ);
      acc0 = wmma_bf16(a0, b0, acc0);
      acc1 = wmma_bf16(a1, b1, acc1);
    }
    const int n_in = lane & 15, mrow = (lane >> 4) * 8;
#pragma unroll
    for (int r = 0; r < 8; ++r) {
      ss[(mt0 * 16 + mrow + r) * LDS_S + nt0 * 16 + n_in] = acc0[r] * 0.125f;
      ss[(mt1 * 16 + mrow + r) * LDS_S + nt1 * 16 + n_in] = acc1[r] * 0.125f;
    }
  }
  __syncthreads();

  // ---- softmax rows (64 rows, thread t<64 owns one row) ----
  if (tid < 64) {
    const float* sr = ss + tid * LDS_S;
    float mx = -1e30f;
    for (int e = 0; e < 64; ++e) mx = fmaxf(mx, sr[e]);
    float sum = 0.f;
    for (int e = 0; e < 64; ++e) sum += __expf(sr[e] - mx);
    float inv = 1.f / sum;
    for (int e = 0; e < 64; ++e)
      ps[tid * LDP + e] = (__bf16)(__expf(sr[e] - mx) * inv);
  }
  __syncthreads();

  // ---- O = P V : 64x256 = 4x16 tiles, 8 per wave, K-dim 64 ----
  {
    const int mtile = wid & 3;
    const int nh    = wid >> 2;
    const f32x8 zero = {0.f, 0.f, 0.f, 0.f, 0.f, 0.f, 0.f, 0.f};
    f32x8 acc[8];
#pragma unroll
    for (int i = 0; i < 8; ++i) acc[i] = zero;
    for (int e0 = 0; e0 < 64; e0 += 32) {
      bf16x16 a = frag_a(ps + (mtile * 16) * LDP + e0, LDP);
#pragma unroll
      for (int nt = 0; nt < 8; ++nt) {
        int ntile = nh * 8 + nt;
        bf16x16 bb = frag_bt(vs + (ntile * 16) * LDV + e0, LDV);
        acc[nt] = wmma_bf16(a, bb, acc[nt]);
      }
    }
    const int n_in = lane & 15, mrow = (lane >> 4) * 8;
#pragma unroll
    for (int nt = 0; nt < 8; ++nt) {
      int ntile = nh * 8 + nt;
      int w = ntile * 16 + n_in;
      int col = h * 64 + (w & 63);
#pragma unroll
      for (int r = 0; r < 8; ++r) {
        int d = mtile * 16 + mrow + r;
        int j = nwin * 256 + d * 4 + (w >> 6);
        o_flat[((size_t)b * L + j) * 1024 + col] = (__bf16)acc[nt][r];
      }
    }
  }
}

// ---------------------------------------------------------------------------
// Small support kernels
// ---------------------------------------------------------------------------
__global__ __launch_bounds__(256) void layernorm_kernel(
    const float* __restrict__ x, const float* __restrict__ g,
    const float* __restrict__ bta, __bf16* __restrict__ out, int C) {
  __shared__ float r1[256], r2[256];
  const int row = blockIdx.x, tid = threadIdx.x;
  const float* xr = x + (size_t)row * C;
  float s = 0.f, ss = 0.f;
  for (int c = tid; c < C; c += 256) { float v = xr[c]; s += v; ss += v * v; }
  r1[tid] = s; r2[tid] = ss; __syncthreads();
  for (int o = 128; o > 0; o >>= 1) {
    if (tid < o) { r1[tid] += r1[tid + o]; r2[tid] += r2[tid + o]; }
    __syncthreads();
  }
  float mean = r1[0] / C;
  float inv = rsqrtf(r2[0] / C - mean * mean + 1e-5f);
  __bf16* orow = out + (size_t)row * C;
  for (int c = tid; c < C; c += 256)
    orow[c] = (__bf16)((xr[c] - mean) * inv * g[c] + bta[c]);
}

__global__ void f32_to_bf16_kernel(const float* __restrict__ s,
                                   __bf16* __restrict__ d, size_t n) {
  size_t i = (size_t)blockIdx.x * blockDim.x + threadIdx.x;
  size_t stride = (size_t)gridDim.x * blockDim.x;
  for (; i < n; i += stride) d[i] = (__bf16)s[i];
}

// conv_w [O,I,3] -> three bf16 [O*I] matrices (tap k at offset k*n)
__global__ void conv_w_split_kernel(const float* __restrict__ s,
                                    __bf16* __restrict__ d, int n) {
  int i = blockIdx.x * blockDim.x + threadIdx.x;
  int stride = gridDim.x * blockDim.x;
  for (; i < n; i += stride) {
    d[i]         = (__bf16)s[(size_t)i * 3 + 0];
    d[n + i]     = (__bf16)s[(size_t)i * 3 + 1];
    d[2 * n + i] = (__bf16)s[(size_t)i * 3 + 2];
  }
}

// MoE gate: one wave per token, 8 experts. gate = softmax(ln3 @ gw^T + gb)
__global__ __launch_bounds__(256) void moe_gate_kernel(
    const __bf16* __restrict__ a, const float* __restrict__ gw,
    const float* __restrict__ gb, float* __restrict__ gate) {
  const int tok = blockIdx.x * 8 + (threadIdx.x >> 5);
  const int lane = threadIdx.x & 31;
  float s[8] = {0.f, 0.f, 0.f, 0.f, 0.f, 0.f, 0.f, 0.f};
  const __bf16* row = a + (size_t)tok * 1024;
  for (int c = lane; c < 1024; c += 32) {
    float xv = (float)row[c];
#pragma unroll
    for (int e = 0; e < 8; ++e) s[e] += xv * gw[e * 1024 + c];
  }
#pragma unroll
  for (int off = 16; off > 0; off >>= 1)
#pragma unroll
    for (int e = 0; e < 8; ++e) s[e] += __shfl_down(s[e], off, 32);
  if (lane == 0) {
    float mx = -1e30f;
#pragma unroll
    for (int e = 0; e < 8; ++e) { s[e] += gb[e]; mx = fmaxf(mx, s[e]); }
    float sum = 0.f;
#pragma unroll
    for (int e = 0; e < 8; ++e) { s[e] = __expf(s[e] - mx); sum += s[e]; }
    float inv = 1.f / sum;
#pragma unroll
    for (int e = 0; e < 8; ++e) gate[(size_t)tok * 8 + e] = s[e] * inv;
  }
}

// ---------------------------------------------------------------------------
// Host orchestration
// ---------------------------------------------------------------------------
template<int MODE>
static void run_gemm(const __bf16* A, const __bf16* Bw, const float* bias,
                     void* C, const float* rs, int sstr, int soff,
                     int M, int N, int K, int shift, int seqlen,
                     hipStream_t st) {
  dim3 g(M / 128, N / 128);
  if (shift == 0)
    gemm_bf16_kernel<MODE, true><<<g, 256, 0, st>>>(
        A, Bw, bias, C, rs, sstr, soff, M, N, K, shift, seqlen);
  else
    gemm_bf16_kernel<MODE, false><<<g, 256, 0, st>>>(
        A, Bw, bias, C, rs, sstr, soff, M, N, K, shift, seqlen);
}

extern "C" void kernel_launch(void* const* d_in, const int* in_sizes, int n_in,
                              void* d_out, int out_size, void* d_ws,
                              size_t ws_size, hipStream_t stream) {
  (void)in_sizes; (void)n_in; (void)out_size; (void)ws_size;
  const int B = 4, L = 2048, C = 1024, H = 16, F = 4096, E = 8;
  const int Ntok = B * L;            // 8192
  const int C3 = 3 * C;

  const float* x_in   = (const float*)d_in[0];
  const float* ln1_g  = (const float*)d_in[1];
  const float* ln1_b  = (const float*)d_in[2];
  const float* qkv_w  = (const float*)d_in[3];
  const float* qkv_b  = (const float*)d_in[4];
  const float* ao_w   = (const float*)d_in[5];
  const float* ao_b   = (const float*)d_in[6];
  const float* ln2_g  = (const float*)d_in[7];
  const float* ln2_b  = (const float*)d_in[8];
  const float* conv_w = (const float*)d_in[9];
  const float* conv_b = (const float*)d_in[10];
  const float* ln3_g  = (const float*)d_in[11];
  const float* ln3_b  = (const float*)d_in[12];
  const float* gate_w = (const float*)d_in[13];
  const float* gate_b = (const float*)d_in[14];
  const float* exp_w  = (const float*)d_in[15];
  const float* exp_b  = (const float*)d_in[16];
  const float* ff1_w  = (const float*)d_in[17];
  const float* ff1_b  = (const float*)d_in[18];
  const float* ff2_w  = (const float*)d_in[19];
  const float* ff2_b  = (const float*)d_in[20];

  float* x = (float*)d_out;  // residual stream lives in d_out (f32)
  char* ws = (char*)d_ws;
  size_t off = 0;
  auto take = [&](size_t bytes) {
    size_t r = off; off += (bytes + 255) & ~(size_t)255; return r;
  };
  __bf16* a_bf   = (__bf16*)(ws + take((size_t)Ntok * C * 2));   // LN/ff input
  char*   bigreg = ws + take((size_t)Ntok * F * 2);              // 64 MB union
  __bf16* qkv_bf = (__bf16*)bigreg;                              // [Ntok,3C]
  __bf16* h_bf   = (__bf16*)bigreg;                              // [Ntok,F]
  __bf16* o_bf   = (__bf16*)(ws + take((size_t)Ntok * C * 2));   // attn o_flat
  float*  gate   = (float*)(ws + take((size_t)Ntok * E * 4));
  __bf16* wqkv   = (__bf16*)(ws + take((size_t)C3 * C * 2));
  __bf16* wao    = (__bf16*)(ws + take((size_t)C * C * 2));
  __bf16* wconv  = (__bf16*)(ws + take((size_t)3 * C * C * 2));
  __bf16* wexp   = (__bf16*)(ws + take((size_t)E * C * C * 2));
  __bf16* wff1   = (__bf16*)(ws + take((size_t)F * C * 2));
  __bf16* wff2   = (__bf16*)(ws + take((size_t)C * F * 2));

  // x <- input
  hipMemcpyAsync(x, x_in, (size_t)Ntok * C * 4, hipMemcpyDeviceToDevice,
                 stream);

  // weights -> bf16
  f32_to_bf16_kernel<<<1024, 256, 0, stream>>>(qkv_w, wqkv, (size_t)C3 * C);
  f32_to_bf16_kernel<<<1024, 256, 0, stream>>>(ao_w, wao, (size_t)C * C);
  conv_w_split_kernel<<<1024, 256, 0, stream>>>(conv_w, wconv, C * C);
  f32_to_bf16_kernel<<<1024, 256, 0, stream>>>(exp_w, wexp, (size_t)E * C * C);
  f32_to_bf16_kernel<<<1024, 256, 0, stream>>>(ff1_w, wff1, (size_t)F * C);
  f32_to_bf16_kernel<<<1024, 256, 0, stream>>>(ff2_w, wff2, (size_t)C * F);

  // ---- attention: x += out_proj(attn(ln1(x))) ----
  layernorm_kernel<<<Ntok, 256, 0, stream>>>(x, ln1_g, ln1_b, a_bf, C);
  run_gemm<0>(a_bf, wqkv, qkv_b, qkv_bf, nullptr, 0, 0, Ntok, C3, C, 0, L,
              stream);
  hipFuncSetAttribute((const void*)attn_window_kernel,
                      hipFuncAttributeMaxDynamicSharedMemorySize, ATTN_SMEM);
  attn_window_kernel<<<dim3(8, H, B), 256, ATTN_SMEM, stream>>>(qkv_bf, o_bf);
  run_gemm<2>(o_bf, wao, ao_b, x, nullptr, 0, 0, Ntok, C, C, 0, L, stream);

  // ---- dilated conv: x += conv(ln2(x)); taps at l-2, l, l+2 ----
  layernorm_kernel<<<Ntok, 256, 0, stream>>>(x, ln2_g, ln2_b, a_bf, C);
  run_gemm<2>(a_bf, wconv,             nullptr, x, nullptr, 0, 0, Ntok, C, C,
              -2, L, stream);
  run_gemm<2>(a_bf, wconv + (size_t)C * C, conv_b, x, nullptr, 0, 0, Ntok, C,
              C, 0, L, stream);
  run_gemm<2>(a_bf, wconv + (size_t)2 * C * C, nullptr, x, nullptr, 0, 0,
              Ntok, C, C, 2, L, stream);

  // ---- MoE: x += sum_e gate[:,e] * (ln3(x) @ We^T + be) ----
  layernorm_kernel<<<Ntok, 256, 0, stream>>>(x, ln3_g, ln3_b, a_bf, C);
  moe_gate_kernel<<<Ntok / 8, 256, 0, stream>>>(a_bf, gate_w, gate_b, gate);
  for (int e = 0; e < E; ++e)
    run_gemm<5>(a_bf, wexp + (size_t)e * C * C, exp_b + (size_t)e * C, x,
                gate, E, e, Ntok, C, C, 0, L, stream);

  // ---- FFN: x += ff2(gelu(ff1(x))) ----
  f32_to_bf16_kernel<<<2048, 256, 0, stream>>>(x, a_bf, (size_t)Ntok * C);
  run_gemm<1>(a_bf, wff1, ff1_b, h_bf, nullptr, 0, 0, Ntok, F, C, 0, L,
              stream);
  run_gemm<2>(h_bf, wff2, ff2_b, x, nullptr, 0, 0, Ntok, C, F, 0, L, stream);
}
